// LinearWithGroupedConv_7696581394667
// MI455X (gfx1250) — matrix operand order repaired and checked by reading, not compiled, not run on hardware
//
#include <hip/hip_runtime.h>
#include <hip/hip_bf16.h>
#include <stdint.h>

// ---------------------------------------------------------------------------
// out[m,n] = sum_k x[m,k] * W[n,k]   (M=8192, N=4096, K=4096, fp32 in/out)
// Strategy: bf16x3 split-precision GEMM on v_wmma_f32_16x16x32_bf16, with
// CDNA5 async global->LDS DMA (GLOBAL_LOAD_ASYNC_TO_LDS_B128 / ASYNCcnt)
// for tile staging.
// ---------------------------------------------------------------------------

#define IN_F   4096
#define OUT_F  4096
#define M_TOT  8192              // 4 * 2048
#define BM     128
#define BN     128
#define BK     32
#define NKT    (IN_F / BK)       // 128 k-iterations

typedef __attribute__((ext_vector_type(16))) __bf16 bf16x16;
typedef __attribute__((ext_vector_type(8)))  float  f32x8;

union Frag {
    bf16x16 v;
    uint4   q[2];
};

__device__ __forceinline__ unsigned short f32_to_bf16_rne(float f) {
    unsigned u = __float_as_uint(f);
    u += 0x7FFFu + ((u >> 16) & 1u);     // round-to-nearest-even
    return (unsigned short)(u >> 16);
}
__device__ __forceinline__ float bf16_to_f32(unsigned short h) {
    return __uint_as_float(((unsigned)h) << 16);
}

// ---------------------------------------------------------------------------
// Pre-pass: fp32 -> (bf16 hi, bf16 lo-residual), vectorized x4.
// ---------------------------------------------------------------------------
__global__ void split_bf16x2_kernel(const float* __restrict__ src,
                                    unsigned short* __restrict__ hi,
                                    unsigned short* __restrict__ lo,
                                    int n4) {
    int i = blockIdx.x * blockDim.x + threadIdx.x;
    if (i >= n4) return;
    const float4 f = reinterpret_cast<const float4*>(src)[i];
    float v[4] = { f.x, f.y, f.z, f.w };
    unsigned short h[4], l[4];
#pragma unroll
    for (int j = 0; j < 4; ++j) {
        h[j] = f32_to_bf16_rne(v[j]);
        l[j] = f32_to_bf16_rne(v[j] - bf16_to_f32(h[j]));
    }
    uint2 hv = make_uint2((unsigned)h[0] | ((unsigned)h[1] << 16),
                          (unsigned)h[2] | ((unsigned)h[3] << 16));
    uint2 lv = make_uint2((unsigned)l[0] | ((unsigned)l[1] << 16),
                          (unsigned)l[2] | ((unsigned)l[3] << 16));
    reinterpret_cast<uint2*>(hi)[i] = hv;
    reinterpret_cast<uint2*>(lo)[i] = lv;
}

// ---------------------------------------------------------------------------
// LDS swizzle: tile part = 128 rows x 32 bf16 (64 B/row, four 16 B chunks).
// Chunk c of row r lives at slot ((c + r) & 3): rotates banks per row while
// keeping every access 16 B aligned.
// ---------------------------------------------------------------------------
__device__ __forceinline__ unsigned lds_off(int row, int chunk) {
    return (unsigned)(row * 64 + (((chunk + row) & 3) << 4));
}

// One async DMA chunk: 32 lanes x 16 B, global (saddr + voffset) -> LDS.
// Tracked on ASYNCcnt; no VGPR staging, no ds_store.
__device__ __forceinline__ void async_chunk(unsigned lds_addr, unsigned voff,
                                            const unsigned short* base) {
    asm volatile("global_load_async_to_lds_b128 %0, %1, %2"
                 :: "v"(lds_addr), "v"(voff),
                    "s"((unsigned long long)(uintptr_t)base)
                 : "memory");
}
__device__ __forceinline__ void wait_asynccnt0() {
    asm volatile("s_wait_asynccnt 0x0" ::: "memory");
}

// A fragment, 16x32 bf16 (ISA 7.12.2): lanes 0-15 -> M=lane, K {0-7,16-23};
// lanes 16-31 -> M=lane-16, K {8-15,24-31}.
__device__ __forceinline__ bf16x16 load_frag_a(const unsigned short* part,
                                               int rowBase, int lane) {
    const char* b = reinterpret_cast<const char*>(part);
    int row  = rowBase + (lane & 15);
    int c0   = (lane >> 4);            // chunk 0 or 1  (K 0-7 / 8-15)
    Frag f;
    f.q[0] = *reinterpret_cast<const uint4*>(b + lds_off(row, c0));
    f.q[1] = *reinterpret_cast<const uint4*>(b + lds_off(row, c0 + 2)); // +16 K
    return f.v;
}

// B fragment, 32x16 bf16: lanes 0-15 -> N=lane, K {0-7,16-23}; lanes 16-31 -> K {8-15,24-31}.
// W tile stored [n][k] in LDS, so chunks are contiguous in K.
__device__ __forceinline__ bf16x16 load_frag_b(const unsigned short* part,
                                               int colBase, int lane) {
    const char* b = reinterpret_cast<const char*>(part);
    int col  = colBase + (lane & 15);
    int c0   = (lane >> 4);            // chunks {0,2} or {1,3}
    Frag f;
    f.q[0] = *reinterpret_cast<const uint4*>(b + lds_off(col, c0));
    f.q[1] = *reinterpret_cast<const uint4*>(b + lds_off(col, c0 + 2));
    return f.v;
}

// ---------------------------------------------------------------------------
// GEMM: 128x128 tile / 256 threads (8 waves, 4x2 wave grid, 32x64 per wave).
// Double-buffered LDS; staging by async DMA overlapped with the WMMA burst.
// ---------------------------------------------------------------------------
__global__ __launch_bounds__(256)
void gemm_bf16x3_kernel(const unsigned short* __restrict__ xh,
                        const unsigned short* __restrict__ xl,
                        const unsigned short* __restrict__ wh,
                        const unsigned short* __restrict__ wl,
                        float* __restrict__ out) {
    // [buf][part: Ah, Al, Bh, Bl][128*32] bf16  -> 64 KB total
    __shared__ __align__(16) unsigned short lds[2][4][BM * BK];

    const int tid  = threadIdx.x;
    const int lane = tid & 31;
    const int wid  = tid >> 5;     // 0..7
    const int wm   = wid & 3;      // wave row 0..3  (32 rows each)
    const int wn   = wid >> 2;     // wave col 0..1  (64 cols each)
    const int blockM = blockIdx.y * BM;
    const int blockN = blockIdx.x * BN;

    // Per-part uniform global bases (k-offset added per iteration, SALU only).
    const unsigned short* partBase[4];
    partBase[0] = xh + (size_t)blockM * IN_F;
    partBase[1] = xl + (size_t)blockM * IN_F;
    partBase[2] = wh + (size_t)blockN * IN_F;
    partBase[3] = wl + (size_t)blockN * IN_F;

    // Per-thread DMA chunk coordinates (computed once).
    // chunk id cid in [0,512): r = cid>>2 (tile row), c = cid&3 (16 B chunk).
    const int r0 = tid >> 2,          c0 = tid & 3;         // j = 0
    const int r1 = (tid + 256) >> 2,  c1 = (tid + 256) & 3; // j = 1
    const unsigned gv0 = (unsigned)(r0 * (IN_F * 2) + c0 * 16); // global byte voffset
    const unsigned gv1 = (unsigned)(r1 * (IN_F * 2) + c1 * 16);
    const unsigned sw0 = lds_off(r0, c0);                   // swizzled LDS offset
    const unsigned sw1 = lds_off(r1, c1);
    const unsigned ldsBase = (unsigned)(uintptr_t)(&lds[0][0][0]);

    f32x8 acc[2][4] = {};          // 2 m-subtiles x 4 n-subtiles

    // Issue all 8 DMA chunks (4 parts x 2) for one k-step into buffer `buf`.
    auto issue_stage = [&](int buf, int k0) {
#pragma unroll
        for (int p = 0; p < 4; ++p) {
            const unsigned short* b = partBase[p] + k0;
            const unsigned lb = ldsBase + (unsigned)buf * 32768u + (unsigned)p * 8192u;
            async_chunk(lb + sw0, gv0, b);
            async_chunk(lb + sw1, gv1, b);
        }
    };

    issue_stage(0, 0);
    wait_asynccnt0();
    __syncthreads();

    for (int kt = 0; kt < NKT; ++kt) {
        const int cur = kt & 1;
        if (kt + 1 < NKT)
            issue_stage(cur ^ 1, (kt + 1) * BK);   // DMA overlaps compute below

        const unsigned short* Ah = lds[cur][0];
        const unsigned short* Al = lds[cur][1];
        const unsigned short* Bh = lds[cur][2];
        const unsigned short* Bl = lds[cur][3];

        bf16x16 ah[2], al[2];
#pragma unroll
        for (int mt = 0; mt < 2; ++mt) {
            ah[mt] = load_frag_a(Ah, wm * 32 + mt * 16, lane);
            al[mt] = load_frag_a(Al, wm * 32 + mt * 16, lane);
        }
#pragma unroll
        for (int nt = 0; nt < 4; ++nt) {
            bf16x16 bh = load_frag_b(Bh, wn * 64 + nt * 16, lane);
            bf16x16 bl = load_frag_b(Bl, wn * 64 + nt * 16, lane);
#pragma unroll
            for (int mt = 0; mt < 2; ++mt) {
                acc[mt][nt] = __builtin_amdgcn_wmma_f32_16x16x32_bf16(
                    false, ah[mt], false, bh, (short)0, acc[mt][nt], false, false);
                acc[mt][nt] = __builtin_amdgcn_wmma_f32_16x16x32_bf16(
                    false, ah[mt], false, bl, (short)0, acc[mt][nt], false, false);
                acc[mt][nt] = __builtin_amdgcn_wmma_f32_16x16x32_bf16(
                    false, al[mt], false, bh, (short)0, acc[mt][nt], false, false);
            }
        }

        if (kt + 1 < NKT) {
            wait_asynccnt0();      // this wave's DMA into lds[cur^1] landed
            __syncthreads();       // everyone's DMA landed; reads of cur done
        }
    }

    // Epilogue: C/D 16x16 f32 layout -> VGPR r, lane<16:(M=r,N=lane),
    // lane>=16:(M=r+8,N=lane-16).
    const int colLane = lane & 15;
    const int rowOff  = (lane >> 4) << 3;   // 0 or 8
#pragma unroll
    for (int mt = 0; mt < 2; ++mt) {
#pragma unroll
        for (int nt = 0; nt < 4; ++nt) {
            const int gr = blockM + wm * 32 + mt * 16 + rowOff;
            const int gc = blockN + wn * 64 + nt * 16 + colLane;
            float* o = out + (size_t)gr * OUT_F + gc;
#pragma unroll
            for (int r = 0; r < 8; ++r)
                o[(size_t)r * OUT_F] = acc[mt][nt][r];
        }
    }
}

// ---------------------------------------------------------------------------
extern "C" void kernel_launch(void* const* d_in, const int* in_sizes, int n_in,
                              void* d_out, int out_size, void* d_ws, size_t ws_size,
                              hipStream_t stream) {
    (void)in_sizes; (void)n_in; (void)out_size; (void)ws_size;

    const float* x = (const float*)d_in[0];      // [4, 2048, 4096]
    const float* w = (const float*)d_in[1];      // [4096, 4096]
    float*     out = (float*)d_out;              // [4, 2048, 4096]

    const size_t xN = (size_t)M_TOT * IN_F;      // 33,554,432
    const size_t wN = (size_t)OUT_F * IN_F;      // 16,777,216

    // Workspace layout (bf16): xh | xl | wh | wl  (192 MiB total)
    unsigned short* xh = (unsigned short*)d_ws;
    unsigned short* xl = xh + xN;
    unsigned short* wh = xl + xN;
    unsigned short* wl = wh + wN;

    const int threads = 256;
    const int xBlocks = (int)((xN / 4 + threads - 1) / threads);
    const int wBlocks = (int)((wN / 4 + threads - 1) / threads);
    split_bf16x2_kernel<<<xBlocks, threads, 0, stream>>>(x, xh, xl, (int)(xN / 4));
    split_bf16x2_kernel<<<wBlocks, threads, 0, stream>>>(w, wh, wl, (int)(wN / 4));

    dim3 grid(OUT_F / BN, M_TOT / BM);           // (32, 64) = 2048 workgroups
    gemm_bf16x3_kernel<<<grid, threads, 0, stream>>>(xh, xl, wh, wl, out);
}
